// twostep_net_37434934952779
// MI455X (gfx1250) — matrix operand order repaired and checked by reading, not compiled, not run on hardware
//
#include <hip/hip_runtime.h>
#include <hip/hip_bf16.h>

typedef __attribute__((ext_vector_type(16))) _Float16 v16h;
typedef __attribute__((ext_vector_type(8)))  _Float16 v8h;
typedef __attribute__((ext_vector_type(8)))  float    v8f;

#define HD 32
#define ED 16
// K = 544 = 16 product chunks of 32 (outer product s (x) a) + 1 chunk of 32 (s alone, bias term)
#define NCHUNK 17
#define BPACK_HALFS (NCHUNK*2*32*16)   // 17408 halfs = 34816 B

// ---------------------------------------------------------------------------
// Pack W'' (544 x 32, f16) into exact B-fragment order:
//   bpack[((c*2+nh)*32 + lane)*16 + j] = W''[c*32 + (j<8 ? (lane>>4)*8 + j : 16 + (lane>>4)*8 + (j-8)), nh*16 + (lane&15)]
// W''[kk,o] = We[(kk/16)*32 + o, kk%16] for kk<512, else be[(kk-512)*32 + o]
// ---------------------------------------------------------------------------
__global__ __launch_bounds__(256) void pack_we_kernel(
    const float* __restrict__ We, const float* __restrict__ be,
    _Float16* __restrict__ bpack)
{
    int idx = blockIdx.x * 256 + threadIdx.x;
    if (idx >= BPACK_HALFS) return;
    int j    = idx & 15;
    int lane = (idx >> 4) & 31;
    int nh   = (idx >> 9) & 1;
    int c    = idx >> 10;
    int half = lane >> 4;
    int col  = nh * 16 + (lane & 15);
    int kk   = c * 32 + (j < 8 ? half * 8 + j : 16 + half * 8 + (j - 8));
    float v;
    if (kk < 512) {
        int i = kk >> 4, k = kk & 15;
        v = We[(i * HD + col) * ED + k];
    } else {
        int i = kk - 512;
        v = be[i * HD + col];
    }
    bpack[idx] = (_Float16)v;
}

// ---------------------------------------------------------------------------
// edge_attr f32 -> f16, once per launch (constant across GRU iterations).
// ---------------------------------------------------------------------------
__global__ __launch_bounds__(256) void cvt_ea_kernel(
    const float* __restrict__ ea, _Float16* __restrict__ ea_h, int n)
{
    int t = blockIdx.x * 256 + threadIdx.x;
    if (t < n) ea_h[t] = (_Float16)ea[t];
}

// ---------------------------------------------------------------------------
// lin0: out = relu(x @ W0^T + b0). One wave per node, lane = output dim.
// Also writes the f16 mirror used by the WMMA edge kernel.
// ---------------------------------------------------------------------------
__global__ __launch_bounds__(256) void lin0_kernel(
    const float* __restrict__ x, const float* __restrict__ W0,
    const float* __restrict__ b0, float* __restrict__ out,
    _Float16* __restrict__ out_h, int N)
{
    __shared__ float lW[32 * 32];
    __shared__ float lb[32];
    for (int i = threadIdx.x; i < 1024; i += 256) lW[i] = W0[i];
    if (threadIdx.x < 32) lb[threadIdx.x] = b0[threadIdx.x];
    __syncthreads();
    int lane = threadIdx.x & 31, wave = threadIdx.x >> 5;
    for (int it = 0; it < 8; ++it) {
        int node = blockIdx.x * 64 + wave * 8 + it;
        if (node >= N) break;
        float xo = x[(size_t)node * HD + lane];
        float acc = lb[lane];
        for (int j = 0; j < 32; ++j)
            acc += __shfl(xo, j, 32) * lW[lane * 32 + j];
        float r = fmaxf(acc, 0.f);
        out[(size_t)node * HD + lane]   = r;
        out_h[(size_t)node * HD + lane] = (_Float16)r;
    }
}

// ---------------------------------------------------------------------------
// Edge message kernel: msg = Z' @ W'' via WMMA f16, scatter-add into agg.
// 4 waves/block, 2 edge-tiles (16 edges) per wave -> 128 edges per block.
//
// A-fragment algebra (16-bit A 16x32 layout; lane: M = lane&15, half = lane>>4):
//   element j in 0..7  : K = j + 8*half          -> s[2c]   * ah[j]
//   element j in 8..15 : K = 16 + (j-8) + 8*half -> s[2c+1] * ah[j-8]
// ah / bias-chunk values are fetched at HALF-DEPENDENT ADDRESSES (16B vector
// loads), so the hot loop has zero cndmask selects and zero f32->f16 cvts:
// just v_pk_mul_f16 + ds_load_b128 + v_wmma.
// ---------------------------------------------------------------------------
__global__ __launch_bounds__(128) void edge_msg_kernel(
    const _Float16* __restrict__ out_h, const int* __restrict__ ei,
    const _Float16* __restrict__ ea_h, const _Float16* __restrict__ bpack,
    float* __restrict__ agg, int E)
{
    __shared__ alignas(32) _Float16 lb[BPACK_HALFS];
    {
        const uint4* g = (const uint4*)bpack;   // 8 halfs per uint4
        uint4* s = (uint4*)lb;
        for (int i = threadIdx.x; i < BPACK_HALFS / 8; i += 128) s[i] = g[i];
    }
    __syncthreads();

    const int lane = threadIdx.x & 31;
    const int wave = threadIdx.x >> 5;
    const int M    = lane & 15;
    const int half = (lane >> 4) & 1;
    const int base = blockIdx.x * 128 + wave * 32;   // 2 tiles: [base, base+31]

    int e  = base + M;        int ec  = (e  < E) ? e  : 0;
    int e2 = base + 16 + M;   int ec2 = (e2 < E) ? e2 : 0;
    int src  = ei[ec];
    int src2 = ei[ec2];
    const _Float16* r0 = out_h + (size_t)src  * HD;
    const _Float16* r1 = out_h + (size_t)src2 * HD;

    // Full source rows (constant-index extracts feed s[2c], s[2c+1]).
    v16h s0lo = *(const v16h*)(r0);
    v16h s0hi = *(const v16h*)(r0 + 16);
    v16h s1lo = *(const v16h*)(r1);
    v16h s1hi = *(const v16h*)(r1 + 16);
    _Float16 s0[32], s1[32];
    #pragma unroll
    for (int i = 0; i < 16; ++i) {
        s0[i] = s0lo[i]; s0[16 + i] = s0hi[i];
        s1[i] = s1lo[i]; s1[16 + i] = s1hi[i];
    }
    // Half-dependent 16B vector loads (no register selects).
    v8h ah0 = *(const v8h*)(ea_h + (size_t)ec  * ED + 8 * half);
    v8h ah1 = *(const v8h*)(ea_h + (size_t)ec2 * ED + 8 * half);
    v8h sh0a = *(const v8h*)(r0 + 8 * half);
    v8h sh0b = *(const v8h*)(r0 + 16 + 8 * half);
    v8h sh1a = *(const v8h*)(r1 + 8 * half);
    v8h sh1b = *(const v8h*)(r1 + 16 + 8 * half);

    // dst indices for the 8 C-fragment rows this lane owns (M = r + 8*half)
    int d0[8], d1[8];
    #pragma unroll
    for (int r = 0; r < 8; ++r) {
        int f0 = base + r + 8 * half;
        int f1 = base + 16 + r + 8 * half;
        d0[r] = (f0 < E) ? ei[E + f0] : -1;
        d1[r] = (f1 < E) ? ei[E + f1] : -1;
    }

    v8f c00 = {}, c01 = {}, c10 = {}, c11 = {};

    // 16 outer-product K-chunks: all register indices compile-time constants.
    #pragma unroll
    for (int c = 0; c < 16; ++c) {
        v16h b0 = *(const v16h*)&lb[((c * 2 + 0) * 32 + lane) * 16];
        v16h b1 = *(const v16h*)&lb[((c * 2 + 1) * 32 + lane) * 16];
        v16h A0, A1;
        #pragma unroll
        for (int j = 0; j < 8; ++j) {
            A0[j]     = s0[2 * c]     * ah0[j];
            A0[8 + j] = s0[2 * c + 1] * ah0[j];
            A1[j]     = s1[2 * c]     * ah1[j];
            A1[8 + j] = s1[2 * c + 1] * ah1[j];
        }
        c00 = __builtin_amdgcn_wmma_f32_16x16x32_f16(false, A0, false, b0, (short)0, c00, false, false);
        c01 = __builtin_amdgcn_wmma_f32_16x16x32_f16(false, A0, false, b1, (short)0, c01, false, false);
        c10 = __builtin_amdgcn_wmma_f32_16x16x32_f16(false, A1, false, b0, (short)0, c10, false, false);
        c11 = __builtin_amdgcn_wmma_f32_16x16x32_f16(false, A1, false, b1, (short)0, c11, false, false);
    }
    // Bias chunk (c == 16): Z'[e, 512+i] = s_i (pairs with packed be rows of W'')
    {
        v16h b0 = *(const v16h*)&lb[((16 * 2 + 0) * 32 + lane) * 16];
        v16h b1 = *(const v16h*)&lb[((16 * 2 + 1) * 32 + lane) * 16];
        v16h A0, A1;
        #pragma unroll
        for (int j = 0; j < 8; ++j) {
            A0[j] = sh0a[j]; A0[8 + j] = sh0b[j];
            A1[j] = sh1a[j]; A1[8 + j] = sh1b[j];
        }
        c00 = __builtin_amdgcn_wmma_f32_16x16x32_f16(false, A0, false, b0, (short)0, c00, false, false);
        c01 = __builtin_amdgcn_wmma_f32_16x16x32_f16(false, A0, false, b1, (short)0, c01, false, false);
        c10 = __builtin_amdgcn_wmma_f32_16x16x32_f16(false, A1, false, b0, (short)0, c10, false, false);
        c11 = __builtin_amdgcn_wmma_f32_16x16x32_f16(false, A1, false, b1, (short)0, c11, false, false);
    }

    // Scatter-add: C VGPR r of lane holds (edge = r + 8*half, odim = lane&15 [+16])
    int od = lane & 15;
    #pragma unroll
    for (int r = 0; r < 8; ++r) {
        if (d0[r] >= 0) {
            atomicAdd(agg + (size_t)d0[r] * HD + od,      c00[r]);
            atomicAdd(agg + (size_t)d0[r] * HD + od + 16, c01[r]);
        }
        if (d1[r] >= 0) {
            atomicAdd(agg + (size_t)d1[r] * HD + od,      c10[r]);
            atomicAdd(agg + (size_t)d1[r] * HD + od + 16, c11[r]);
        }
    }
}

// ---------------------------------------------------------------------------
// Node update: m = relu(agg + out@Wroot^T + bconv); out = GRU(m, out).
// One wave per node, lane = feature dim; weights staged in LDS.
// ---------------------------------------------------------------------------
__global__ __launch_bounds__(256) void node_update_kernel(
    float* __restrict__ out, _Float16* __restrict__ out_h,
    const float* __restrict__ agg,
    const float* __restrict__ Wroot, const float* __restrict__ bconv,
    const float* __restrict__ Wih, const float* __restrict__ bih,
    const float* __restrict__ Whh, const float* __restrict__ bhh, int N)
{
    __shared__ float lWr[32 * 32], lWi[96 * 32], lWh[96 * 32];
    __shared__ float lbc[32], lbi[96], lbh[96];
    for (int i = threadIdx.x; i < 1024; i += 256) lWr[i] = Wroot[i];
    for (int i = threadIdx.x; i < 3072; i += 256) { lWi[i] = Wih[i]; lWh[i] = Whh[i]; }
    if (threadIdx.x < 32) lbc[threadIdx.x] = bconv[threadIdx.x];
    if (threadIdx.x < 96) { lbi[threadIdx.x] = bih[threadIdx.x]; lbh[threadIdx.x] = bhh[threadIdx.x]; }
    __syncthreads();

    int lane = threadIdx.x & 31, wave = threadIdx.x >> 5;
    for (int it = 0; it < 8; ++it) {
        int node = blockIdx.x * 64 + wave * 8 + it;
        if (node >= N) break;
        float fo = out[(size_t)node * HD + lane];       // h == out entering the step
        float root = 0.f;
        for (int j = 0; j < 32; ++j)
            root += __shfl(fo, j, 32) * lWr[lane * 32 + j];
        float mo = fmaxf(agg[(size_t)node * HD + lane] + root + lbc[lane], 0.f);

        float gir = lbi[lane], giz = lbi[32 + lane], gin = lbi[64 + lane];
        float ghr = lbh[lane], ghz = lbh[32 + lane], ghn = lbh[64 + lane];
        for (int j = 0; j < 32; ++j) {
            float mj = __shfl(mo, j, 32);
            float hj = __shfl(fo, j, 32);
            gir += mj * lWi[lane * 32 + j];
            giz += mj * lWi[(32 + lane) * 32 + j];
            gin += mj * lWi[(64 + lane) * 32 + j];
            ghr += hj * lWh[lane * 32 + j];
            ghz += hj * lWh[(32 + lane) * 32 + j];
            ghn += hj * lWh[(64 + lane) * 32 + j];
        }
        float rg = 1.f / (1.f + __expf(-(gir + ghr)));
        float zg = 1.f / (1.f + __expf(-(giz + ghz)));
        float ng = tanhf(gin + rg * ghn);
        float hnew = (1.f - zg) * ng + zg * fo;
        out[(size_t)node * HD + lane]   = hnew;
        out_h[(size_t)node * HD + lane] = (_Float16)hnew;
    }
}

// ---------------------------------------------------------------------------
// Global mean pool (accumulate) + final linear.
// ---------------------------------------------------------------------------
__global__ __launch_bounds__(256) void pool_kernel(
    const float* __restrict__ out, const int* __restrict__ batch,
    float* __restrict__ sums, float* __restrict__ cnt, int N)
{
    int t = blockIdx.x * 256 + threadIdx.x;
    if (t >= N * HD) return;
    int n = t >> 5, d = t & 31;
    int g = batch[n];
    atomicAdd(&sums[(size_t)g * HD + d], out[t]);
    if (d == 0) atomicAdd(&cnt[g], 1.f);
}

__global__ __launch_bounds__(256) void finalize_kernel(
    const float* __restrict__ sums, const float* __restrict__ cnt,
    const float* __restrict__ W1, const float* __restrict__ b1,
    float* __restrict__ y, int G)
{
    int lane = threadIdx.x & 31, wave = threadIdx.x >> 5;
    int g = blockIdx.x * 8 + wave;
    if (g >= G) return;
    float p = sums[(size_t)g * HD + lane] / fmaxf(cnt[g], 1.f);
    float v = p * W1[lane];
    for (int off = 16; off > 0; off >>= 1) v += __shfl_down(v, off, 32);
    if (lane == 0) y[g] = v + b1[0];
}

// ---------------------------------------------------------------------------
extern "C" void kernel_launch(void* const* d_in, const int* in_sizes, int n_in,
                              void* d_out, int out_size, void* d_ws, size_t ws_size,
                              hipStream_t stream) {
    const float* x     = (const float*)d_in[0];
    const int*   ei    = (const int*)  d_in[1];
    const float* ea    = (const float*)d_in[2];
    const int*   batch = (const int*)  d_in[3];
    const float* W0    = (const float*)d_in[4];
    const float* b0    = (const float*)d_in[5];
    const float* We    = (const float*)d_in[6];
    const float* be    = (const float*)d_in[7];
    const float* Wroot = (const float*)d_in[8];
    const float* bconv = (const float*)d_in[9];
    const float* Wih   = (const float*)d_in[10];
    const float* bih   = (const float*)d_in[11];
    const float* Whh   = (const float*)d_in[12];
    const float* bhh   = (const float*)d_in[13];
    const float* W1    = (const float*)d_in[14];
    const float* b1    = (const float*)d_in[15];
    float* y = (float*)d_out;

    const int N = in_sizes[0] / HD;
    const int E = in_sizes[1] / 2;
    const int G = out_size;

    char* w = (char*)d_ws;
    size_t off = 0;
    auto carve = [&](size_t bytes) { size_t r = off; off = (off + bytes + 255) & ~(size_t)255; return r; };
    float*     out   = (float*)(w + carve((size_t)N * HD * 4));
    float*     agg   = (float*)(w + carve((size_t)N * HD * 4));
    float*     sums  = (float*)(w + carve((size_t)G * HD * 4));
    float*     cnt   = (float*)(w + carve((size_t)G * 4));
    _Float16*  bpack = (_Float16*)(w + carve((size_t)BPACK_HALFS * 2));
    _Float16*  out_h = (_Float16*)(w + carve((size_t)N * HD * 2));
    _Float16*  ea_h  = (_Float16*)(w + carve((size_t)E * ED * 2));
    (void)ws_size; (void)n_in;

    pack_we_kernel<<<(BPACK_HALFS + 255) / 256, 256, 0, stream>>>(We, be, bpack);
    cvt_ea_kernel<<<((size_t)E * ED + 255) / 256, 256, 0, stream>>>(ea, ea_h, E * ED);
    lin0_kernel<<<(N + 63) / 64, 256, 0, stream>>>(x, W0, b0, out, out_h, N);

    for (int it = 0; it < 3; ++it) {
        hipMemsetAsync(agg, 0, (size_t)N * HD * 4, stream);
        edge_msg_kernel<<<(E + 127) / 128, 128, 0, stream>>>(out_h, ei, ea_h, bpack, agg, E);
        node_update_kernel<<<(N + 63) / 64, 256, 0, stream>>>(out, out_h, agg, Wroot, bconv,
                                                              Wih, bih, Whh, bhh, N);
    }

    hipMemsetAsync(sums, 0, (size_t)G * HD * 4, stream);
    hipMemsetAsync(cnt, 0, (size_t)G * 4, stream);
    pool_kernel<<<((size_t)N * HD + 255) / 256, 256, 0, stream>>>(out, batch, sums, cnt, N);
    finalize_kernel<<<(G + 7) / 8, 256, 0, stream>>>(sums, cnt, W1, b1, y, G);
}
